// NonLocalAttModule_86938728005778
// MI455X (gfx1250) — matrix operand order repaired and checked by reading, not compiled, not run on hardware
//
#include <hip/hip_runtime.h>

// ---------------- problem constants ----------------
constexpr int BB_  = 2;
constexpr int HH_  = 80;
constexpr int WW_  = 80;
constexpr int NF_  = 128;   // n_filter (C)
constexpr int CIN_ = 256;   // 2*NF
constexpr int NN_  = HH_ * WW_;  // 6400

typedef unsigned short u16;
typedef __attribute__((ext_vector_type(16))) __bf16 v16bf;
typedef __attribute__((ext_vector_type(8)))  float  v8f;
typedef __attribute__((ext_vector_type(8)))  u16    u16x8;   // one 128-bit load

union BF16x16 {
    v16bf    v;
    u16      s[16];
    unsigned u[8];
    u16x8    h[2];
};
union F32x8 { v8f v; float f[8]; };

// round-to-nearest-even f32 -> bf16 bits
__device__ __forceinline__ u16 f2bfbits(float f) {
    union { float f; unsigned u; } c; c.f = f;
    unsigned u = c.u;
    return (u16)((u + 0x7FFFu + ((u >> 16) & 1u)) >> 16);
}

// 16-bit A/B fragment (16x32 / 32x16) per ISA 7.12.2.
// For a fixed half-wave, the 16 fragment elements are two contiguous
// 8-element runs in K: [half*8 .. half*8+7] and [16+half*8 .. 16+half*8+7].
// -> each fragment = two 128-bit loads from (base + half*8) and (base + 16 + half*8).
__device__ __forceinline__ void load_frag128(BF16x16& f, const u16* base, int half) {
    f.h[0] = *(const u16x8*)(base + half * 8);
    f.h[1] = *(const u16x8*)(base + 16 + half * 8);
}

// ---------------- f32 -> bf16 convert ----------------
__global__ void f32_to_bf16_k(const float* __restrict__ in, u16* __restrict__ out, int n) {
    int i = blockIdx.x * 256 + threadIdx.x;
    if (i < n) out[i] = f2bfbits(in[i]);
}

// ---------------- implicit-GEMM 3x3 SAME conv ----------------
// One wave: 16 consecutive pixels (one row segment) x 16 output filters.
// K loop: 9 taps x (CIN_T/32) chunks of v_wmma_f32_16x16x32_bf16.
template <int CIN_T, bool RESID>
__global__ void __launch_bounds__(32)
conv3x3_wmma(const u16* __restrict__ in,
             const u16* __restrict__ wt, int cout,          // HWIO (3,3,CIN_T,cout)
             u16* __restrict__ out_bf,                      // bf16 out (if !RESID)
             const float* __restrict__ resid,               // residual (if RESID)
             float* __restrict__ out_f)                     // f32 out  (if RESID)
{
    const int lane = threadIdx.x & 31;
    const int half = lane >> 4;
    const int mr   = lane & 15;

    const int WT = WW_ / 16;                 // 5 tiles per row
    const int bx = blockIdx.x;
    const int xt = bx % WT;
    const int yy = (bx / WT) % HH_;
    const int bb = bx / (WT * HH_);
    const int x0 = xt * 16;
    const int f0 = blockIdx.y * 16;

    F32x8 acc;
#pragma unroll
    for (int r = 0; r < 8; ++r) acc.f[r] = 0.0f;

    const int kk = mr + 16 * half;

    for (int t = 0; t < 9; ++t) {
        const int dy = t / 3 - 1, dx = t % 3 - 1;
        const int py = yy + dy;
        const int px = x0 + mr + dx;
        const bool ok = (py >= 0) & (py < HH_) & (px >= 0) & (px < WW_);
        // boundary: clamp address (always valid), zero via bitwise mask (no branches)
        const unsigned msk = ok ? 0xFFFFFFFFu : 0u;
        const int pyc = ok ? py : 0;
        const int pxc = ok ? px : 0;
        const u16* srow = in + (((size_t)bb * HH_ + pyc) * WW_ + pxc) * CIN_T;

#pragma unroll
        for (int kc = 0; kc < CIN_T; kc += 32) {
            BF16x16 a, w;
            load_frag128(a, srow + kc, half);
#pragma unroll
            for (int i = 0; i < 8; ++i) a.u[i] &= msk;      // v_and, EXEC untouched

            const u16* wrow = wt + ((size_t)t * CIN_T + kc) * cout + f0 + mr;
#pragma unroll
            for (int s = 0; s < 16; ++s) {
                const int k = (s < 8) ? (8 * half + s) : (16 + 8 * half + s - 8);
                w.s[s] = wrow[(size_t)k * cout];
            }

            acc.v = __builtin_amdgcn_wmma_f32_16x16x32_bf16(
                false, a.v, false, w.v, (short)0, acc.v, false, false);
        }
    }

    const size_t p0 = ((size_t)bb * HH_ + yy) * WW_ + x0;
#pragma unroll
    for (int r = 0; r < 8; ++r) {
        const size_t idx = (p0 + r + 8 * half) * cout + f0 + mr;
        if (RESID) out_f[idx] = resid[idx] + acc.f[r];
        else       out_bf[idx] = f2bfbits(acc.f[r]);
    }
}

// ---------------- flash attention (softmax(theta.phi).g) ----------------
// One wave handles 16 query rows. Two passes over m (rowmax, then exp+PV),
// S tiles via WMMA; P tile staged through LDS to re-layout C-frag -> A-frag.
__global__ void __launch_bounds__(32)
flash_attn_wmma(const u16* __restrict__ th,   // (B,N,C) bf16
                const u16* __restrict__ ph,   // raw (B,N*C) bf16; viewed (B,C,N) row-major
                const u16* __restrict__ gg,   // (B,N,C) bf16
                u16* __restrict__ outp)       // (B,N,C) bf16
{
    const int lane = threadIdx.x & 31;
    const int half = lane >> 4;
    const int mr   = lane & 15;
    const int bb   = blockIdx.y;
    const int n0   = blockIdx.x * 16;
    const int C    = NF_;

    __shared__ u16 pbuf[16 * 32];

    // preload theta A-fragments: 16 rows x 128 K, as 4 fragments of 16x32
    BF16x16 ta[4];
    const u16* trow = th + ((size_t)bb * NN_ + n0 + mr) * C;
#pragma unroll
    for (int j = 0; j < 4; ++j) load_frag128(ta[j], trow + j * 32, half);

    const int kk = mr + 16 * half;
    const u16* phbase = ph + (size_t)bb * NN_ * C;

    // ---- pass 1: row max ----
    float rowmax[8];
#pragma unroll
    for (int r = 0; r < 8; ++r) rowmax[r] = -3.0e38f;

    for (int m0 = 0; m0 < NN_; m0 += 32) {
        F32x8 s0, s1;
#pragma unroll
        for (int r = 0; r < 8; ++r) { s0.f[r] = 0.0f; s1.f[r] = 0.0f; }

#pragma unroll
        for (int ks = 0; ks < 4; ++ks) {
            const u16* pb = phbase + (size_t)(ks * 32) * NN_ + m0 + mr;
            __builtin_prefetch(pb + 64, 0, 1);
            BF16x16 b0, b1;
#pragma unroll
            for (int s = 0; s < 16; ++s) {
                const int k = (s < 8) ? (8 * half + s) : (16 + 8 * half + s - 8);
                b0.s[s] = pb[(size_t)k * NN_];
                b1.s[s] = pb[(size_t)k * NN_ + 16];
            }
            s0.v = __builtin_amdgcn_wmma_f32_16x16x32_bf16(
                false, ta[ks].v, false, b0.v, (short)0, s0.v, false, false);
            s1.v = __builtin_amdgcn_wmma_f32_16x16x32_bf16(
                false, ta[ks].v, false, b1.v, (short)0, s1.v, false, false);
        }
#pragma unroll
        for (int r = 0; r < 8; ++r) {
            float lm = fmaxf(s0.f[r], s1.f[r]);
#pragma unroll
            for (int d = 1; d < 16; d <<= 1) lm = fmaxf(lm, __shfl_xor(lm, d, 16));
            rowmax[r] = fmaxf(rowmax[r], lm);
        }
    }

    // ---- pass 2: exp, row sum, O += P.g ----
    float rowsum[8];
    F32x8 o[8];
#pragma unroll
    for (int r = 0; r < 8; ++r) rowsum[r] = 0.0f;
#pragma unroll
    for (int cg = 0; cg < 8; ++cg)
#pragma unroll
        for (int r = 0; r < 8; ++r) o[cg].f[r] = 0.0f;

    for (int m0 = 0; m0 < NN_; m0 += 32) {
        F32x8 s0, s1;
#pragma unroll
        for (int r = 0; r < 8; ++r) { s0.f[r] = 0.0f; s1.f[r] = 0.0f; }

#pragma unroll
        for (int ks = 0; ks < 4; ++ks) {
            const u16* pb = phbase + (size_t)(ks * 32) * NN_ + m0 + mr;
            BF16x16 b0, b1;
#pragma unroll
            for (int s = 0; s < 16; ++s) {
                const int k = (s < 8) ? (8 * half + s) : (16 + 8 * half + s - 8);
                b0.s[s] = pb[(size_t)k * NN_];
                b1.s[s] = pb[(size_t)k * NN_ + 16];
            }
            s0.v = __builtin_amdgcn_wmma_f32_16x16x32_bf16(
                false, ta[ks].v, false, b0.v, (short)0, s0.v, false, false);
            s1.v = __builtin_amdgcn_wmma_f32_16x16x32_bf16(
                false, ta[ks].v, false, b1.v, (short)0, s1.v, false, false);
        }

        // exponentiate, stage P into LDS (C-layout), reduce row sums
#pragma unroll
        for (int r = 0; r < 8; ++r) {
            float e0 = __expf(s0.f[r] - rowmax[r]);
            float e1 = __expf(s1.f[r] - rowmax[r]);
            pbuf[(r + 8 * half) * 32 + mr]      = f2bfbits(e0);
            pbuf[(r + 8 * half) * 32 + 16 + mr] = f2bfbits(e1);
            float ps = e0 + e1;
#pragma unroll
            for (int d = 1; d < 16; d <<= 1) ps += __shfl_xor(ps, d, 16);
            rowsum[r] += ps;
        }
        __syncthreads();

        // reload P as A-fragment (16 rows x 32 K): two ds_load_b128 per lane
        BF16x16 pa;
        load_frag128(pa, &pbuf[mr * 32], half);
        __syncthreads();

        // O[16 x 128] += P[16 x 32] . g[32 x 128]
        const u16* gs = gg + ((size_t)bb * NN_ + m0) * C + mr;
        __builtin_prefetch(gs + 32 * C, 0, 1);
#pragma unroll
        for (int cg = 0; cg < 8; ++cg) {
            BF16x16 gbf;
#pragma unroll
            for (int s = 0; s < 16; ++s) {
                const int k = (s < 8) ? (8 * half + s) : (16 + 8 * half + s - 8);
                gbf.s[s] = gs[(size_t)k * C + cg * 16];
            }
            o[cg].v = __builtin_amdgcn_wmma_f32_16x16x32_bf16(
                false, pa.v, false, gbf.v, (short)0, o[cg].v, false, false);
        }
    }

    // normalize and store bf16
#pragma unroll
    for (int cg = 0; cg < 8; ++cg)
#pragma unroll
        for (int r = 0; r < 8; ++r) {
            const size_t idx = ((size_t)bb * NN_ + n0 + r + 8 * half) * C + cg * 16 + mr;
            outp[idx] = f2bfbits(o[cg].f[r] / rowsum[r]);
        }
}

// ---------------- host launcher ----------------
extern "C" void kernel_launch(void* const* d_in, const int* in_sizes, int n_in,
                              void* d_out, int out_size, void* d_ws, size_t ws_size,
                              hipStream_t stream) {
    const float* x   = (const float*)d_in[0];
    const float* wth = (const float*)d_in[1];
    const float* wph = (const float*)d_in[2];
    const float* wg  = (const float*)d_in[3];
    const float* wcv = (const float*)d_in[4];
    float* out = (float*)d_out;

    // workspace layout (bf16 / u16 elements)
    u16* p = (u16*)d_ws;
    u16* xb   = p; p += (size_t)BB_ * NN_ * CIN_;   // x in bf16
    u16* wthb = p; p += (size_t)9 * CIN_ * NF_;
    u16* wphb = p; p += (size_t)9 * CIN_ * NF_;
    u16* wgb  = p; p += (size_t)9 * CIN_ * NF_;
    u16* wcvb = p; p += (size_t)9 * NF_ * CIN_;
    u16* thb  = p; p += (size_t)BB_ * NN_ * NF_;    // theta bf16
    u16* phb  = p; p += (size_t)BB_ * NN_ * NF_;    // phi   bf16 (raw conv layout)
    u16* gb   = p; p += (size_t)BB_ * NN_ * NF_;    // g     bf16
    u16* ab   = p; p += (size_t)BB_ * NN_ * NF_;    // attention output bf16

    auto cvt = [&](const float* src, u16* dst, int n) {
        f32_to_bf16_k<<<(n + 255) / 256, 256, 0, stream>>>(src, dst, n);
    };
    cvt(x,   xb,   BB_ * NN_ * CIN_);
    cvt(wth, wthb, 9 * CIN_ * NF_);
    cvt(wph, wphb, 9 * CIN_ * NF_);
    cvt(wg,  wgb,  9 * CIN_ * NF_);
    cvt(wcv, wcvb, 9 * NF_ * CIN_);

    // theta / phi / g convs: 256 -> 128, bf16 outputs
    dim3 gq(BB_ * HH_ * (WW_ / 16), NF_ / 16);          // (800, 8)
    conv3x3_wmma<CIN_, false><<<gq, 32, 0, stream>>>(xb, wthb, NF_, thb, nullptr, nullptr);
    conv3x3_wmma<CIN_, false><<<gq, 32, 0, stream>>>(xb, wphb, NF_, phb, nullptr, nullptr);
    conv3x3_wmma<CIN_, false><<<gq, 32, 0, stream>>>(xb, wgb,  NF_, gb,  nullptr, nullptr);

    // attention: softmax(theta . phi) . g
    dim3 ga(NN_ / 16, BB_);                              // (400, 2)
    flash_attn_wmma<<<ga, 32, 0, stream>>>(thb, phb, gb, ab);

    // final conv 128 -> 256 with residual add, fp32 output
    dim3 gf(BB_ * HH_ * (WW_ / 16), CIN_ / 16);          // (800, 16)
    conv3x3_wmma<NF_, true><<<gf, 32, 0, stream>>>(ab, wcvb, CIN_, nullptr, x, out);
}
